// MultiHeadAttention_59253368815938
// MI455X (gfx1250) — matrix operand order repaired and checked by reading, not compiled, not run on hardware
//
#include <hip/hip_runtime.h>
#include <hip/hip_bf16.h>

typedef __attribute__((ext_vector_type(16))) _Float16 v16h;
typedef __attribute__((ext_vector_type(8)))  _Float16 v8h;
typedef __attribute__((ext_vector_type(4)))  _Float16 v4h;
typedef __attribute__((ext_vector_type(8)))  float    v8f;
typedef __attribute__((ext_vector_type(4)))  float    v4f;

#define B_  4
#define S_  2048
#define E_  1024
#define H_  16
#define DH_ 64
#define M1_ (B_*S_)   /* 8192 rows of x */

// ---------------------------------------------------------------------------
// WMMA helpers (CDNA5: D = A(16x32 f16) * B(32x16 f16) + C(16x16 f32))
// ---------------------------------------------------------------------------
__device__ __forceinline__ v8f wmma16(const v16h& a, const v16h& b, const v8f& c) {
  return __builtin_amdgcn_wmma_f32_16x16x32_f16(
      /*neg_a=*/false, a, /*neg_b=*/false, b,
      /*c_mod=*/(short)0, c, /*reuse_a=*/false, /*reuse_b=*/false);
}

__device__ __forceinline__ v8f zero8() {
  v8f z;
#pragma unroll
  for (int i = 0; i < 8; ++i) z[i] = 0.0f;
  return z;
}

// Generic LDS pointer -> 32-bit LDS byte address (low bits of flat address)
__device__ __forceinline__ unsigned lds_off(const void* p) {
  return (unsigned)(unsigned long long)p;
}

// Async global->LDS copy of 16 bytes per lane (ASYNCcnt-tracked, no VGPRs).
__device__ __forceinline__ void async_ld_b128(unsigned lds_byte_addr,
                                              const void* gaddr) {
  asm volatile("global_load_async_to_lds_b128 %0, %1, off"
               :: "v"(lds_byte_addr), "v"(gaddr) : "memory");
}
__device__ __forceinline__ void wait_async0() {
  asm volatile("s_wait_asynccnt 0x0" ::: "memory");
}

// A fragment (16x32 f16). lane L holds row M=L%16;
// halves 0..7  -> K = kk*32 + (L>=16 ? 8:0) + 0..7
// halves 8..15 -> K = kk*32 + (L>=16 ? 8:0) + 16..23
__device__ __forceinline__ v16h ld_afrag(const _Float16* lds, int row, int stride,
                                         int kk, int lane) {
  const int kbase = kk * 32 + ((lane >> 4) ? 8 : 0);
  const v8h lo = *(const v8h*)&lds[row * stride + kbase];
  const v8h hi = *(const v8h*)&lds[row * stride + kbase + 16];
  v16h a;
#pragma unroll
  for (int i = 0; i < 8; ++i) { a[i] = lo[i]; a[8 + i] = hi[i]; }
  return a;
}

// B fragment (32x16 f16). lane L holds col N=L%16;
// halves 0..7  -> K = kk*32 + (L>=16 ? 8:0) + 0..7; halves 8..15 -> +16..23   (LDS stored [n][k])
__device__ __forceinline__ v16h ld_bfrag(const _Float16* lds, int col, int stride,
                                         int kk, int lane) {
  const int kbase = kk * 32 + ((lane >> 4) ? 8 : 0);
  const v8h lo = *(const v8h*)&lds[col * stride + kbase];
  const v8h hi = *(const v8h*)&lds[col * stride + kbase + 16];
  v16h b;
#pragma unroll
  for (int i = 0; i < 8; ++i) { b[i] = lo[i]; b[8 + i] = hi[i]; }
  return b;
}

// ---------------------------------------------------------------------------
// Kernel 1: fused QKV projection.
// grid = (M1/64, H, 3), block = 128 (4 waves). 64x64 C tile, K-step 64,
// double-buffered LDS (one barrier per K-step, fills overlap WMMA).
// ---------------------------------------------------------------------------
__global__ __launch_bounds__(128) void qkv_kernel(
    const float* __restrict__ x,
    const float* __restrict__ Wq, const float* __restrict__ bq,
    const float* __restrict__ Wk, const float* __restrict__ bk,
    const float* __restrict__ Wv, const float* __restrict__ bv,
    _Float16* __restrict__ qws, _Float16* __restrict__ kws,
    _Float16* __restrict__ vws) {
  const int which = blockIdx.z;
  const float* W    = (which == 0) ? Wq : (which == 1) ? Wk : Wv;
  const float* bias = (which == 0) ? bq : (which == 1) ? bk : bv;
  _Float16* out     = (which == 0) ? qws : (which == 1) ? kws : vws;

  const int h     = blockIdx.y;
  const int mBase = blockIdx.x * 64;
  const int tid   = threadIdx.x;
  const int lane  = tid & 31, wave = tid >> 5;
  const int ln    = lane & 15, lhalf = lane >> 4;

  __shared__ _Float16 AtB[2][64 * 72];  // [m][k]
  __shared__ _Float16 BtB[2][64 * 72];  // [n][k] transposed

  const size_t wbase = (size_t)h * E_ * DH_;

  auto fill = [&](int buf, int k0) {
    // A tile: 64x64 fp32 -> f16, float4 loads, packed 8B LDS stores.
#pragma unroll
    for (int it = 0; it < 8; ++it) {            // 64 rows * 16 chunks / 128
      const int i = tid + it * 128;
      const int r = i >> 4, c = (i & 15) * 4;
      const v4f f = *(const v4f*)&x[(size_t)(mBase + r) * E_ + k0 + c];
      v4h hlf;
#pragma unroll
      for (int j = 0; j < 4; ++j) hlf[j] = (_Float16)f[j];
      *(v4h*)&AtB[buf][r * 72 + c] = hlf;
    }
    // B tile (transposed): Bt[n][k] = W[h][k0+k][n], float4 along n.
#pragma unroll
    for (int it = 0; it < 8; ++it) {            // 64 k * 16 chunks / 128
      const int i = tid + it * 128;
      const int k = i >> 4, n0 = (i & 15) * 4;
      const v4f f = *(const v4f*)&W[wbase + (size_t)(k0 + k) * DH_ + n0];
#pragma unroll
      for (int j = 0; j < 4; ++j) BtB[buf][(n0 + j) * 72 + k] = (_Float16)f[j];
    }
  };

  v8f acc[4];
#pragma unroll
  for (int i = 0; i < 4; ++i) acc[i] = zero8();

  fill(0, 0);
  __syncthreads();

  for (int ks = 0; ks < 16; ++ks) {             // 16 K-steps of 64 over E
    const int cur = ks & 1, nxt = cur ^ 1;
    if (ks + 1 < 16) fill(nxt, (ks + 1) * 64);  // overlaps with WMMA below

    const _Float16* A  = AtB[cur];
    const _Float16* Bm = BtB[cur];
    const int row = wave * 16 + ln;
    const v16h a0  = ld_afrag(A, row, 72, 0, lane);
    const v16h a1  = ld_afrag(A, row, 72, 1, lane);
    const v16h b00 = ld_bfrag(Bm,  0 + ln, 72, 0, lane);
    const v16h b10 = ld_bfrag(Bm, 16 + ln, 72, 0, lane);
    const v16h b20 = ld_bfrag(Bm, 32 + ln, 72, 0, lane);
    const v16h b30 = ld_bfrag(Bm, 48 + ln, 72, 0, lane);
    const v16h b01 = ld_bfrag(Bm,  0 + ln, 72, 1, lane);
    const v16h b11 = ld_bfrag(Bm, 16 + ln, 72, 1, lane);
    const v16h b21 = ld_bfrag(Bm, 32 + ln, 72, 1, lane);
    const v16h b31 = ld_bfrag(Bm, 48 + ln, 72, 1, lane);
    acc[0] = wmma16(a0, b00, acc[0]);
    acc[1] = wmma16(a0, b10, acc[1]);
    acc[2] = wmma16(a0, b20, acc[2]);
    acc[3] = wmma16(a0, b30, acc[3]);
    acc[0] = wmma16(a1, b01, acc[0]);
    acc[1] = wmma16(a1, b11, acc[1]);
    acc[2] = wmma16(a1, b21, acc[2]);
    acc[3] = wmma16(a1, b31, acc[3]);

    __syncthreads();
  }

  // Epilogue: add bias, store f16 [B,H,S,Dh]
#pragma unroll
  for (int sub = 0; sub < 4; ++sub) {
    const float bval = bias[h * DH_ + sub * 16 + ln];
#pragma unroll
    for (int r = 0; r < 8; ++r) {
      const int m = mBase + wave * 16 + r + lhalf * 8;
      const int bb = m / S_, s = m % S_;
      out[(((size_t)bb * H_ + h) * S_ + s) * DH_ + sub * 16 + ln] =
          (_Float16)(acc[sub][r] + bval);
    }
  }
}

// ---------------------------------------------------------------------------
// Kernel 2: flash attention over one (b,h), 64-query block per workgroup.
// grid = (S/64, B*H), block = 128 (4 waves, each owns 16 query rows).
// Double-buffered K/V tiles; K arrives via async global->LDS.
// ---------------------------------------------------------------------------
__global__ __launch_bounds__(128) void attn_kernel(
    const _Float16* __restrict__ qws, const _Float16* __restrict__ kws,
    const _Float16* __restrict__ vws, _Float16* __restrict__ ows) {
  const int bh    = blockIdx.y;
  const int qBase = blockIdx.x * 64;
  const int tid   = threadIdx.x, lane = tid & 31, wave = tid >> 5;
  const int ln    = lane & 15, lhalf = lane >> 4;
  const float kScale = 0.125f;  // 1/sqrt(Dh)

  __shared__ _Float16 Qt[64 * 72];         // [m][d]
  __shared__ _Float16 KtB[2][64 * 72];     // [t][d]
  __shared__ _Float16 VtB[2][64 * 72];     // [d][t] transposed
  __shared__ _Float16 Pt[64 * 72];         // [m][t] per-wave strips

  const _Float16* qp = qws + (size_t)bh * S_ * DH_;
  const _Float16* kp = kws + (size_t)bh * S_ * DH_;
  const _Float16* vp = vws + (size_t)bh * S_ * DH_;

  auto fillKV = [&](int buf, int t0) {
    // K tile [t][d]: async straight 16B copies
#pragma unroll
    for (int it = 0; it < 4; ++it) {             // 64 rows * 8 chunks / 128
      const int i = tid + it * 128;
      const int t = i >> 3, c = (i & 7) * 8;
      async_ld_b128(lds_off(&KtB[buf][t * 72 + c]),
                    &kp[(size_t)(t0 + t) * DH_ + c]);
    }
    // V tile transposed [d][t]: vector load + scatter
#pragma unroll
    for (int it = 0; it < 4; ++it) {
      const int i = tid + it * 128;
      const int t = i >> 3, d0 = (i & 7) * 8;
      const v8h tmp = *(const v8h*)&vp[(size_t)(t0 + t) * DH_ + d0];
#pragma unroll
      for (int j = 0; j < 8; ++j) VtB[buf][(d0 + j) * 72 + t] = tmp[j];
    }
  };

  // Resident Q tile: async copies; first tile of K/V.
#pragma unroll
  for (int it = 0; it < 4; ++it) {
    const int i = tid + it * 128;
    const int r = i >> 3, c = (i & 7) * 8;
    async_ld_b128(lds_off(&Qt[r * 72 + c]), &qp[(size_t)(qBase + r) * DH_ + c]);
  }
  fillKV(0, 0);
  wait_async0();
  __syncthreads();

  // Q A-fragments are loop-invariant: hoist.
  const int row = wave * 16 + ln;
  const v16h qa0 = ld_afrag(Qt, row, 72, 0, lane);
  const v16h qa1 = ld_afrag(Qt, row, 72, 1, lane);

  v8f oacc[4];
  float mrow[8], lrow[8];
#pragma unroll
  for (int i = 0; i < 4; ++i) oacc[i] = zero8();
#pragma unroll
  for (int r = 0; r < 8; ++r) { mrow[r] = -1e30f; lrow[r] = 0.0f; }

  for (int kb = 0; kb < S_ / 64; ++kb) {
    const int cur = kb & 1, nxt = cur ^ 1;
    if (kb + 1 < S_ / 64) fillKV(nxt, (kb + 1) * 64);  // overlaps compute

    const _Float16* Kt = KtB[cur];
    const _Float16* Vt = VtB[cur];

    // S = Q * K^T
    v8f sc[4];
#pragma unroll
    for (int i = 0; i < 4; ++i) sc[i] = zero8();
    {
      const v16h b00 = ld_bfrag(Kt,  0 + ln, 72, 0, lane);
      const v16h b10 = ld_bfrag(Kt, 16 + ln, 72, 0, lane);
      const v16h b20 = ld_bfrag(Kt, 32 + ln, 72, 0, lane);
      const v16h b30 = ld_bfrag(Kt, 48 + ln, 72, 0, lane);
      const v16h b01 = ld_bfrag(Kt,  0 + ln, 72, 1, lane);
      const v16h b11 = ld_bfrag(Kt, 16 + ln, 72, 1, lane);
      const v16h b21 = ld_bfrag(Kt, 32 + ln, 72, 1, lane);
      const v16h b31 = ld_bfrag(Kt, 48 + ln, 72, 1, lane);
      sc[0] = wmma16(qa0, b00, sc[0]);
      sc[1] = wmma16(qa0, b10, sc[1]);
      sc[2] = wmma16(qa0, b20, sc[2]);
      sc[3] = wmma16(qa0, b30, sc[3]);
      sc[0] = wmma16(qa1, b01, sc[0]);
      sc[1] = wmma16(qa1, b11, sc[1]);
      sc[2] = wmma16(qa1, b21, sc[2]);
      sc[3] = wmma16(qa1, b31, sc[3]);
    }

    // Online softmax over this wave's 16 rows.
    float p[4][8];
#pragma unroll
    for (int r = 0; r < 8; ++r) {
      float v = -1e30f;
#pragma unroll
      for (int sub = 0; sub < 4; ++sub) v = fmaxf(v, sc[sub][r]);
#pragma unroll
      for (int msk = 1; msk < 16; msk <<= 1) v = fmaxf(v, __shfl_xor(v, msk, 16));
      const float sm   = v * kScale;
      const float mnew = fmaxf(mrow[r], sm);
      const float sOld = __expf(mrow[r] - mnew);
      mrow[r] = mnew;
      float rsum = 0.0f;
#pragma unroll
      for (int sub = 0; sub < 4; ++sub) {
        const float pe = __expf(sc[sub][r] * kScale - mnew);
        p[sub][r] = pe;
        rsum += pe;
      }
#pragma unroll
      for (int msk = 1; msk < 16; msk <<= 1) rsum += __shfl_xor(rsum, msk, 16);
      lrow[r] = lrow[r] * sOld + rsum;
#pragma unroll
      for (int sub = 0; sub < 4; ++sub) oacc[sub][r] *= sOld;
    }

    // Spill P into A-fragment layout (per-wave strip; same-wave LDS in order).
#pragma unroll
    for (int sub = 0; sub < 4; ++sub)
#pragma unroll
      for (int r = 0; r < 8; ++r)
        Pt[(wave * 16 + r + lhalf * 8) * 72 + sub * 16 + ln] = (_Float16)p[sub][r];

    // O += P * V
    {
      const v16h a0  = ld_afrag(Pt, row, 72, 0, lane);
      const v16h a1  = ld_afrag(Pt, row, 72, 1, lane);
      const v16h b00 = ld_bfrag(Vt,  0 + ln, 72, 0, lane);
      const v16h b10 = ld_bfrag(Vt, 16 + ln, 72, 0, lane);
      const v16h b20 = ld_bfrag(Vt, 32 + ln, 72, 0, lane);
      const v16h b30 = ld_bfrag(Vt, 48 + ln, 72, 0, lane);
      const v16h b01 = ld_bfrag(Vt,  0 + ln, 72, 1, lane);
      const v16h b11 = ld_bfrag(Vt, 16 + ln, 72, 1, lane);
      const v16h b21 = ld_bfrag(Vt, 32 + ln, 72, 1, lane);
      const v16h b31 = ld_bfrag(Vt, 48 + ln, 72, 1, lane);
      oacc[0] = wmma16(a0, b00, oacc[0]);
      oacc[1] = wmma16(a0, b10, oacc[1]);
      oacc[2] = wmma16(a0, b20, oacc[2]);
      oacc[3] = wmma16(a0, b30, oacc[3]);
      oacc[0] = wmma16(a1, b01, oacc[0]);
      oacc[1] = wmma16(a1, b11, oacc[1]);
      oacc[2] = wmma16(a1, b21, oacc[2]);
      oacc[3] = wmma16(a1, b31, oacc[3]);
    }

    wait_async0();     // our async deposits into nxt are done
    __syncthreads();   // everyone's deposits visible; cur free for reuse
  }

  // Normalize and store O as f16 [B,H,S,Dh]
  _Float16* ob = ows + (size_t)bh * S_ * DH_;
#pragma unroll
  for (int sub = 0; sub < 4; ++sub)
#pragma unroll
    for (int r = 0; r < 8; ++r) {
      const int m = qBase + wave * 16 + r + lhalf * 8;
      ob[(size_t)m * DH_ + sub * 16 + ln] = (_Float16)(oacc[sub][r] / lrow[r]);
    }
}

// ---------------------------------------------------------------------------
// Kernel 3: output projection  out[m][n] = sum_k O[m][k]*Wo[k][n] + bo[n]
// A gathered (async) from [B,H,S,Dh] f16; double-buffered; fp32 epilogue.
// grid = (M1/64, E/64), block = 128.
// ---------------------------------------------------------------------------
__global__ __launch_bounds__(128) void oproj_kernel(
    const _Float16* __restrict__ ows, const float* __restrict__ Wo,
    const float* __restrict__ bo, float* __restrict__ out) {
  const int nBase = blockIdx.y * 64;
  const int mBase = blockIdx.x * 64;
  const int tid   = threadIdx.x, lane = tid & 31, wave = tid >> 5;
  const int ln    = lane & 15, lhalf = lane >> 4;

  __shared__ _Float16 AtB[2][64 * 72];
  __shared__ _Float16 BtB[2][64 * 72];

  auto fill = [&](int buf, int k0) {
    const int hh = k0 >> 6;                     // k-tile sits inside one head
    // A tile: async f16 copy from [B,H,S,Dh]
#pragma unroll
    for (int it = 0; it < 4; ++it) {            // 64 rows * 8 chunks / 128
      const int i = tid + it * 128;
      const int r = i >> 3, c = (i & 7) * 8;
      const int m = mBase + r;
      const int bb = m / S_, s = m % S_;
      async_ld_b128(lds_off(&AtB[buf][r * 72 + c]),
                    &ows[(((size_t)bb * H_ + hh) * S_ + s) * DH_ + c]);
    }
    // B tile transposed: Bt[n][k] = Wo[k0+k][nBase+n]
#pragma unroll
    for (int it = 0; it < 8; ++it) {            // 64 k * 16 chunks / 128
      const int i = tid + it * 128;
      const int k = i >> 4, n0 = (i & 15) * 4;
      const v4f f = *(const v4f*)&Wo[(size_t)(k0 + k) * E_ + nBase + n0];
#pragma unroll
      for (int j = 0; j < 4; ++j) BtB[buf][(n0 + j) * 72 + k] = (_Float16)f[j];
    }
  };

  v8f acc[4];
#pragma unroll
  for (int i = 0; i < 4; ++i) acc[i] = zero8();

  fill(0, 0);
  wait_async0();
  __syncthreads();

  for (int ks = 0; ks < 16; ++ks) {
    const int cur = ks & 1, nxt = cur ^ 1;
    if (ks + 1 < 16) fill(nxt, (ks + 1) * 64);

    const _Float16* A  = AtB[cur];
    const _Float16* Bm = BtB[cur];
    const int row = wave * 16 + ln;
    const v16h a0  = ld_afrag(A, row, 72, 0, lane);
    const v16h a1  = ld_afrag(A, row, 72, 1, lane);
    const v16h b00 = ld_bfrag(Bm,  0 + ln, 72, 0, lane);
    const v16h b10 = ld_bfrag(Bm, 16 + ln, 72, 0, lane);
    const v16h b20 = ld_bfrag(Bm, 32 + ln, 72, 0, lane);
    const v16h b30 = ld_bfrag(Bm, 48 + ln, 72, 0, lane);
    const v16h b01 = ld_bfrag(Bm,  0 + ln, 72, 1, lane);
    const v16h b11 = ld_bfrag(Bm, 16 + ln, 72, 1, lane);
    const v16h b21 = ld_bfrag(Bm, 32 + ln, 72, 1, lane);
    const v16h b31 = ld_bfrag(Bm, 48 + ln, 72, 1, lane);
    acc[0] = wmma16(a0, b00, acc[0]);
    acc[1] = wmma16(a0, b10, acc[1]);
    acc[2] = wmma16(a0, b20, acc[2]);
    acc[3] = wmma16(a0, b30, acc[3]);
    acc[0] = wmma16(a1, b01, acc[0]);
    acc[1] = wmma16(a1, b11, acc[1]);
    acc[2] = wmma16(a1, b21, acc[2]);
    acc[3] = wmma16(a1, b31, acc[3]);

    wait_async0();
    __syncthreads();
  }

#pragma unroll
  for (int sub = 0; sub < 4; ++sub) {
    const float bval = bo[nBase + sub * 16 + ln];
#pragma unroll
    for (int r = 0; r < 8; ++r) {
      const int m = mBase + wave * 16 + r + lhalf * 8;
      out[(size_t)m * E_ + nBase + sub * 16 + ln] = acc[sub][r] + bval;
    }
  }
}

// ---------------------------------------------------------------------------
extern "C" void kernel_launch(void* const* d_in, const int* in_sizes, int n_in,
                              void* d_out, int out_size, void* d_ws, size_t ws_size,
                              hipStream_t stream) {
  (void)in_sizes; (void)n_in; (void)out_size; (void)ws_size;
  const float* x  = (const float*)d_in[0];
  const float* Wq = (const float*)d_in[1];
  const float* bq = (const float*)d_in[2];
  const float* Wk = (const float*)d_in[3];
  const float* bk = (const float*)d_in[4];
  const float* Wv = (const float*)d_in[5];
  const float* bv = (const float*)d_in[6];
  const float* Wo = (const float*)d_in[7];
  const float* bo = (const float*)d_in[8];
  float* out = (float*)d_out;

  const size_t per = (size_t)B_ * H_ * S_ * DH_;  // 8,388,608 f16 elements
  _Float16* qws = (_Float16*)d_ws;
  _Float16* kws = qws + per;
  _Float16* vws = kws + per;
  _Float16* ows = vws + per;

  const dim3 blk(128, 1, 1);
  qkv_kernel<<<dim3(M1_ / 64, H_, 3), blk, 0, stream>>>(x, Wq, bq, Wk, bk, Wv, bv,
                                                        qws, kws, vws);
  attn_kernel<<<dim3(S_ / 64, B_ * H_), blk, 0, stream>>>(qws, kws, vws, ows);
  oproj_kernel<<<dim3(M1_ / 64, E_ / 64), blk, 0, stream>>>(ows, Wo, bo, out);
}
